// Model_14121852470088
// MI455X (gfx1250) — matrix operand order repaired and checked by reading, not compiled, not run on hardware
//
#include <hip/hip_runtime.h>
#include <hip/hip_bf16.h>
#include <math.h>

// MI455X / gfx1250 implementation. All large GEMMs + attention run on
// v_wmma_f32_16x16x32_f16 (f16 in, f32 accumulate); GEMM tiles are staged
// into LDS by the Tensor Data Mover (tensor_load_to_lds + s_wait_tensorcnt),
// double-buffered one K-slab ahead.  Requires ~238 MB d_ws.

typedef __attribute__((ext_vector_type(16))) _Float16 v16h;
typedef __attribute__((ext_vector_type(8)))  _Float16 v8h;
typedef __attribute__((ext_vector_type(8)))  float    v8f;
typedef __attribute__((ext_vector_type(4)))  unsigned int v4u;
typedef __attribute__((ext_vector_type(8)))  int          v8i;
typedef __attribute__((ext_vector_type(4)))  int          v4i;

#define EMB    768
#define HEADS  2
#define DHEAD  384
#define BATCH  32
#define TXT    77
#define NTOK   654            // 1 + 576 + 77 real tokens
#define NP     672            // padded to 21*32 for exact WMMA tiling
#define NPATCH 576

#if __has_builtin(__builtin_amdgcn_tensor_load_to_lds) && __has_builtin(__builtin_amdgcn_s_wait_tensorcnt)
#define USE_TDM 1
#endif

__device__ __forceinline__ v16h cat8(v8h lo, v8h hi) {
  return __builtin_shufflevector(lo, hi, 0,1,2,3,4,5,6,7,8,9,10,11,12,13,14,15);
}
__device__ __forceinline__ v8f wmma_f16(v16h a, v16h b, v8f c) {
  return __builtin_amdgcn_wmma_f32_16x16x32_f16(false, a, false, b, (short)0, c, false, false);
}

#ifdef USE_TDM
// TDM descriptor for a 128-row x 32-half f16 tile of a [rows][768] f16 tensor.
// pad_interval = 16 DWORDs (one 64B row), pad_amount = 4 DWORDs (16B)
// -> LDS row stride 80B, exactly matching the [128][40] padded layout.
// 6-arg builtin form (clang-23 / therock): (g0, g1, g2, g3, g4, cpol).
__device__ __forceinline__ void tdm_load_tile(unsigned lds_off,
                                              const _Float16* gsrc,
                                              unsigned rows_total) {
  unsigned long long ga = (unsigned long long)(size_t)gsrc;
  v4u g0;
  g0[0] = 1u;                                         // count=1, user mode
  g0[1] = lds_off;                                    // lds_addr (bytes)
  g0[2] = (unsigned)(ga & 0xFFFFFFFFu);               // global_addr[31:0]
  g0[3] = (unsigned)((ga >> 32) & 0x01FFFFFFu) | (2u << 30);  // addr[56:32], type=2
  v8i g1;
  g1[0] = (int)((1u << 16)    // data_size = 2 bytes
              | (1u << 20)    // pad_enable
              | (3u << 22)    // pad_interval: 16 DWORDs
              | (3u << 25));  // pad_amount: 4 DWORDs
  g1[1] = (int)(768u << 16);                          // tensor_dim0 = 768 (lo16)
  g1[2] = (int)((rows_total & 0xFFFFu) << 16);        // dim0 hi=0 | tensor_dim1 lo16
  g1[3] = (int)(((rows_total >> 16) & 0xFFFFu) | (32u << 16)); // dim1 hi | tile_dim0=32
  g1[4] = (int)128;                                   // tile_dim1=128, tile_dim2=0
  g1[5] = (int)768;                                   // tensor_dim0_stride lo32
  g1[6] = 0;                                          // stride hi | dim1_stride lo
  g1[7] = 0;
  v4i g2; g2[0] = 1; g2[1] = 1; g2[2] = 0; g2[3] = 0; // dim2=1, dim3=1 (unused)
  v4i g3; g3[0] = 0; g3[1] = (int)(1u << 16); g3[2] = 0; g3[3] = 0; // dim4=1
  v8i g4 = {};                                        // unused trailing group
  __builtin_amdgcn_tensor_load_to_lds(g0, g1, g2, g3, g4, 0);
}
#endif

// ---------------- weight transpose + f16 convert:  wt[n][k] = (f16)w[k][n] ----
__global__ __launch_bounds__(256) void k_w_t16(const float* __restrict__ w,
                                               _Float16* __restrict__ wt) {
  int id = blockIdx.x * 256 + threadIdx.x;      // 768*768
  int n = id / EMB, kk = id % EMB;
  wt[id] = (_Float16)w[(size_t)kk * EMB + n];
}

// ---------------- im2col + f16 convert: patches [B*576, 768] -----------------
__global__ __launch_bounds__(256) void k_im2col(const float* __restrict__ x,
                                                _Float16* __restrict__ p16) {
  size_t id = (size_t)blockIdx.x * 256 + threadIdx.x;   // 32*576*768
  int j = (int)(id % EMB);
  size_t r = id / EMB;
  int pidx = (int)(r % NPATCH);
  int b    = (int)(r / NPATCH);
  int ph = pidx / 24, pw = pidx % 24;
  int p1 = j / 48; int r2 = j % 48; int p2 = r2 / 3; int c = r2 % 3;
  p16[id] = (_Float16)x[(((size_t)(b * 3 + c) * 384) + ph * 16 + p1) * 384 + pw * 16 + p2];
}

// ---------------- build t: cls row, text rows, zero pad rows -----------------
__global__ __launch_bounds__(256) void k_init_t(const float* __restrict__ text,
                                                const float* __restrict__ cls,
                                                float* __restrict__ t) {
  int row = blockIdx.x;              // B*NP rows
  int b = row / NP, i = row % NP;
  if (i >= 1 && i <= NPATCH) return;               // patch GEMM fills these
  for (int e = threadIdx.x; e < EMB; e += 256) {
    float v;
    if (i == 0)            v = cls[e];
    else if (i < NTOK)     v = text[((size_t)b * TXT + (i - 577)) * EMB + e];
    else                   v = 0.f;
    t[(size_t)row * EMB + e] = v;
  }
}

// ---------------- WMMA GEMM: Y[M,768] = A16[M,768] @ W + bias ----------------
// Bt is the pre-transposed f16 weight: Bt[n][k] = W[k][n].
enum { MODE_PATCH = 0, MODE_QKV = 1, MODE_VT = 2, MODE_PLAIN = 3 };

template <int MODE>
__global__ __launch_bounds__(256) void k_gemm(const _Float16* __restrict__ A,
                                              const _Float16* __restrict__ Bt,
                                              const float* __restrict__ bias,
                                              void* __restrict__ out,
                                              int Mtot) {
  __shared__ _Float16 As[2][128][40];   // +8 halves pad -> conflict-free frags
  __shared__ _Float16 Bs[2][128][40];   // (TDM reproduces pad via pad_amount)
  const int tid  = threadIdx.x;
  const int lane = tid & 31, wave = tid >> 5;
  const int wm = wave & 3, wn = wave >> 2;       // 4 waves in M, 2 in N
  const size_t m0 = (size_t)blockIdx.x * 128;
  const int    n0 = blockIdx.y * 128;

  v8f acc[2][4] = {};
  const int ar   = lane & 15;
  const int aoff = (lane < 16) ? 0 : 16;
  const int boff = (lane < 16) ? 0 : 16;

#ifdef USE_TDM
  if (wave == 0) {   // TDM issues per-wave (EXEC ignored): one wave drives DMA
    tdm_load_tile((unsigned)(size_t)&As[0][0][0], A + m0 * EMB, (unsigned)Mtot);
    tdm_load_tile((unsigned)(size_t)&Bs[0][0][0], Bt + (size_t)n0 * EMB, (unsigned)EMB);
  }
#endif
  int buf = 0;
  for (int k0 = 0; k0 < EMB; k0 += 32) {
    __syncthreads();                   // everyone done reading buf^1
#ifdef USE_TDM
    if (wave == 0) {
      if (k0 + 32 < EMB) {             // prefetch next K-slab into other buffer
        tdm_load_tile((unsigned)(size_t)&As[buf ^ 1][0][0],
                      A + m0 * EMB + k0 + 32, (unsigned)Mtot);
        tdm_load_tile((unsigned)(size_t)&Bs[buf ^ 1][0][0],
                      Bt + (size_t)n0 * EMB + k0 + 32, (unsigned)EMB);
        __builtin_amdgcn_s_wait_tensorcnt((short)2);  // current pair complete
      } else {
        __builtin_amdgcn_s_wait_tensorcnt((short)0);
      }
    }
#else
    for (int s = tid; s < 512; s += 256) {       // manual staging fallback
      int row = s >> 2, c8 = s & 3;
      *(v8h*)&As[buf][row][c8 * 8] = *(const v8h*)(A  + (m0 + row) * EMB + k0 + c8 * 8);
      *(v8h*)&Bs[buf][row][c8 * 8] = *(const v8h*)(Bt + (size_t)(n0 + row) * EMB + k0 + c8 * 8);
    }
#endif
    __syncthreads();                   // tile visible to all waves
    v16h af[2];
    for (int i = 0; i < 2; i++) {
      const _Float16* p = &As[buf][wm * 32 + i * 16 + ar][0];
      af[i] = cat8(*(const v8h*)(p + aoff), *(const v8h*)(p + aoff + 8));
    }
    for (int j = 0; j < 4; j++) {
      const _Float16* p = &Bs[buf][wn * 64 + j * 16 + ar][boff];
      v16h bf_ = cat8(*(const v8h*)p, *(const v8h*)(p + 8));
      for (int i = 0; i < 2; i++) acc[i][j] = wmma_f16(af[i], bf_, acc[i][j]);
    }
    buf ^= 1;
  }

  // epilogue: bias + scatter per MODE.  C layout: VGPR j -> row j (+8 for hi half)
  const int lr  = (lane < 16) ? 0 : 8;
  const int ln_ = lane & 15;
  for (int i = 0; i < 2; i++) {
    for (int j = 0; j < 8; j++) {
      size_t r = m0 + wm * 32 + i * 16 + j + lr;
      size_t b_ = 0, ii = 0, trow = r;
      if (MODE == MODE_PATCH) { b_ = r / NPATCH; ii = r % NPATCH; trow = b_ * NP + 1 + ii; }
      if (MODE == MODE_QKV || MODE == MODE_VT) { b_ = r / NP; ii = r % NP; }
      for (int jn = 0; jn < 4; jn++) {
        int col = n0 + wn * 64 + jn * 16 + ln_;
        float val = acc[i][jn][j] + bias[col];
        if (MODE == MODE_PATCH) {
          ((float*)out)[trow * EMB + col] = val;
        } else if (MODE == MODE_PLAIN) {
          ((float*)out)[r * EMB + col] = val;
        } else {
          int head = col / DHEAD, dc = col - head * DHEAD;
          if (MODE == MODE_QKV)
            ((_Float16*)out)[(((size_t)(b_ * HEADS + head) * NP) + ii) * DHEAD + dc] = (_Float16)val;
          else  // MODE_VT: store V transposed [bh][D][NP]
            ((_Float16*)out)[(((size_t)(b_ * HEADS + head) * DHEAD) + dc) * NP + ii] = (_Float16)val;
        }
      }
    }
  }
}

// ---------------- LayerNorm kernels ------------------------------------------
__global__ __launch_bounds__(256) void k_ln1(const float* __restrict__ t,
                                             const float* __restrict__ g,
                                             const float* __restrict__ be,
                                             _Float16* __restrict__ h16) {
  int row = blockIdx.x, tid = threadIdx.x;
  int i = row % NP;
  if (i >= NTOK) {                                   // zero pad rows
    for (int e = tid; e < EMB; e += 256) h16[(size_t)row * EMB + e] = (_Float16)0.f;
    return;
  }
  __shared__ float red[256];
  const float* tr = t + (size_t)row * EMB;
  float x0 = tr[tid], x1 = tr[tid + 256], x2 = tr[tid + 512];
  red[tid] = x0 + x1 + x2; __syncthreads();
  for (int s = 128; s > 0; s >>= 1) { if (tid < s) red[tid] += red[tid + s]; __syncthreads(); }
  float mean = red[0] * (1.f / EMB); __syncthreads();
  float d0 = x0 - mean, d1 = x1 - mean, d2 = x2 - mean;
  red[tid] = d0 * d0 + d1 * d1 + d2 * d2; __syncthreads();
  for (int s = 128; s > 0; s >>= 1) { if (tid < s) red[tid] += red[tid + s]; __syncthreads(); }
  float rstd = rsqrtf(red[0] * (1.f / EMB) + 1e-5f);
  _Float16* hr = h16 + (size_t)row * EMB;
  hr[tid]       = (_Float16)(d0 * rstd * g[tid]       + be[tid]);
  hr[tid + 256] = (_Float16)(d1 * rstd * g[tid + 256] + be[tid + 256]);
  hr[tid + 512] = (_Float16)(d2 * rstd * g[tid + 512] + be[tid + 512]);
}

__global__ __launch_bounds__(256) void k_ln2res(const float* __restrict__ o,
                                                const float* __restrict__ g,
                                                const float* __restrict__ be,
                                                float* __restrict__ t) {
  int row = blockIdx.x, tid = threadIdx.x;
  __shared__ float red[256];
  const float* orow = o + (size_t)row * EMB;
  float x0 = orow[tid], x1 = orow[tid + 256], x2 = orow[tid + 512];
  red[tid] = x0 + x1 + x2; __syncthreads();
  for (int s = 128; s > 0; s >>= 1) { if (tid < s) red[tid] += red[tid + s]; __syncthreads(); }
  float mean = red[0] * (1.f / EMB); __syncthreads();
  float d0 = x0 - mean, d1 = x1 - mean, d2 = x2 - mean;
  red[tid] = d0 * d0 + d1 * d1 + d2 * d2; __syncthreads();
  for (int s = 128; s > 0; s >>= 1) { if (tid < s) red[tid] += red[tid + s]; __syncthreads(); }
  float rstd = rsqrtf(red[0] * (1.f / EMB) + 1e-5f);
  float* trw = t + (size_t)row * EMB;
  trw[tid]       += d0 * rstd * g[tid]       + be[tid];
  trw[tid + 256] += d1 * rstd * g[tid + 256] + be[tid + 256];
  trw[tid + 512] += d2 * rstd * g[tid + 512] + be[tid + 512];
}

__global__ __launch_bounds__(256) void k_lnrow(const float* __restrict__ in,
                                               const float* __restrict__ g,
                                               const float* __restrict__ be,
                                               float* __restrict__ outp) {
  int row = blockIdx.x, tid = threadIdx.x;
  __shared__ float red[256];
  const float* ir = in + (size_t)row * EMB;
  float x0 = ir[tid], x1 = ir[tid + 256], x2 = ir[tid + 512];
  red[tid] = x0 + x1 + x2; __syncthreads();
  for (int s = 128; s > 0; s >>= 1) { if (tid < s) red[tid] += red[tid + s]; __syncthreads(); }
  float mean = red[0] * (1.f / EMB); __syncthreads();
  float d0 = x0 - mean, d1 = x1 - mean, d2 = x2 - mean;
  red[tid] = d0 * d0 + d1 * d1 + d2 * d2; __syncthreads();
  for (int s = 128; s > 0; s >>= 1) { if (tid < s) red[tid] += red[tid + s]; __syncthreads(); }
  float rstd = rsqrtf(red[0] * (1.f / EMB) + 1e-5f);
  float* orow = outp + (size_t)row * EMB;
  orow[tid]       = d0 * rstd * g[tid]       + be[tid];
  orow[tid + 256] = d1 * rstd * g[tid + 256] + be[tid + 256];
  orow[tid + 512] = d2 * rstd * g[tid + 512] + be[tid + 512];
}

// ---------------- flash-style WMMA attention ---------------------------------
// q,k: [B*H][NP][384] f16 ; v: [B*H][384][NP] f16 (transposed) ;
// ao: [B][NP][768] f16 with col = h*384 + d.  softmax then *1/sqrt(768).
__global__ __launch_bounds__(256, 1) void k_attn(const _Float16* __restrict__ q,
                                                 const _Float16* __restrict__ k,
                                                 const _Float16* __restrict__ v,
                                                 _Float16* __restrict__ ao) {
  __shared__ _Float16 ks[32][392];    // 32 keys x 384 d (+8 pad)
  __shared__ _Float16 vTs[384][40];   // 384 d x 32 keys (+8 pad)
  __shared__ _Float16 ps[8][16][32];  // per-wave P transpose patch
  const int tid = threadIdx.x;
  const int lane = tid & 31, wave = tid >> 5;
  const int bh = blockIdx.y;
  const int b = bh / HEADS, h = bh % HEADS;
  const int q0 = blockIdx.x * 128 + wave * 16;   // this wave's 16 query rows
  const size_t qkBase = (size_t)bh * NP * DHEAD;
  const size_t vBase  = (size_t)bh * DHEAD * NP;

  const int rl   = lane & 15;
  const int aoff = (lane < 16) ? 0 : 16;
  const int boff = (lane < 16) ? 0 : 16;

  // preload Q fragments from global (reused for all 21 K-tiles)
  int grow = q0 + rl; if (grow >= NP) grow = NP - 1;
  v16h aq[12];
  for (int c = 0; c < 12; c++) {
    const _Float16* p = q + qkBase + (size_t)grow * DHEAD + c * 32;
    aq[c] = cat8(*(const v8h*)(p + aoff), *(const v8h*)(p + aoff + 8));
  }

  v8f oacc[24] = {};
  float mrow[8], lrow[8];
  for (int j = 0; j < 8; j++) { mrow[j] = -1e30f; lrow[j] = 0.f; }

  for (int it = 0; it < NP / 32; ++it) {
    const int kt0 = it * 32;
    __syncthreads();
    for (int s = tid; s < 1536; s += 256) {     // stage K tile (row-major)
      int row = s / 48, c8 = s % 48;
      *(v8h*)&ks[row][c8 * 8] = *(const v8h*)(k + qkBase + (size_t)(kt0 + row) * DHEAD + c8 * 8);
    }
    for (int s = tid; s < 1536; s += 256) {     // stage V tile (d-major)
      int d = s >> 2, c8 = s & 3;
      *(v8h*)&vTs[d][c8 * 8] = *(const v8h*)(v + vBase + (size_t)d * NP + kt0 + c8 * 8);
    }
    __syncthreads();

    // S = Q . K^T  (two 16x16 score tiles over 12 d-chunks)
    v8f s0 = {}, s1 = {};
    for (int c = 0; c < 12; c++) {
      const _Float16* p0 = &ks[rl][c * 32 + boff];
      const _Float16* p1 = &ks[16 + rl][c * 32 + boff];
      v16h b0 = cat8(*(const v8h*)p0, *(const v8h*)(p0 + 8));
      v16h b1 = cat8(*(const v8h*)p1, *(const v8h*)(p1 + 8));
      s0 = wmma_f16(aq[c], b0, s0);
      s1 = wmma_f16(aq[c], b1, s1);
    }
    if (kt0 + rl >= NTOK)       for (int j = 0; j < 8; j++) s0[j] = -1e30f;
    if (kt0 + 16 + rl >= NTOK)  for (int j = 0; j < 8; j++) s1[j] = -1e30f;

    // online softmax (row reductions across the 16-lane half-wave)
    const int prow_hi = (lane < 16) ? 0 : 8;
    for (int j = 0; j < 8; j++) {
      float a0 = s0[j], a1 = s1[j];
      float mx = fmaxf(a0, a1);
      for (int off = 1; off < 16; off <<= 1) mx = fmaxf(mx, __shfl_xor(mx, off, 32));
      float mnew = fmaxf(mrow[j], mx);
      float corr = __expf(mrow[j] - mnew);
      float p0 = __expf(a0 - mnew);
      float p1 = __expf(a1 - mnew);
      float psum = p0 + p1;
      for (int off = 1; off < 16; off <<= 1) psum += __shfl_xor(psum, off, 32);
      lrow[j] = lrow[j] * corr + psum;
      mrow[j] = mnew;
      ps[wave][j + prow_hi][rl]      = (_Float16)p0;
      ps[wave][j + prow_hi][16 + rl] = (_Float16)p1;
      for (int c = 0; c < 24; c++) oacc[c][j] *= corr;   // rescale running O
    }
    __syncthreads();

    // O += P . V   (P transposed via per-wave LDS patch into A fragment)
    const int poff = (lane < 16) ? 0 : 16;
    const _Float16* pp = &ps[wave][rl][poff];
    v16h ap = cat8(*(const v8h*)pp, *(const v8h*)(pp + 8));
    for (int c = 0; c < 24; c++) {
      const _Float16* pv = &vTs[c * 16 + rl][boff];
      v16h bv_ = cat8(*(const v8h*)pv, *(const v8h*)(pv + 8));
      oacc[c] = wmma_f16(ap, bv_, oacc[c]);
    }
  }

  // normalize (post-softmax 1/sqrt(EMB) scale, faithful to reference) + store
  const float sc = 0.03608439182435161f;   // 1/sqrt(768)
  float inv[8];
  for (int j = 0; j < 8; j++) inv[j] = sc / lrow[j];
  for (int j = 0; j < 8; j++) {
    int gr = q0 + j + ((lane < 16) ? 0 : 8);
    if (gr >= NP) continue;
    size_t base = ((size_t)b * NP + gr) * EMB + h * DHEAD;
    for (int c = 0; c < 24; c++)
      ao[base + c * 16 + rl] = (_Float16)(oacc[c][j] * inv[j]);
  }
}

// ---------------- pool / final linear ----------------------------------------
__global__ __launch_bounds__(256) void k_pool(const float* __restrict__ t,
                                              float* __restrict__ pooled) {
  int id = blockIdx.x * 256 + threadIdx.x;     // B*768
  int b = id / EMB, e = id % EMB;
  const float* p = t + (size_t)b * NP * EMB + e;
  float s = 0.f;
  for (int i = 0; i < NTOK; i++) s += p[(size_t)i * EMB];
  pooled[id] = s * (1.f / NTOK);
}

__global__ __launch_bounds__(256) void k_final(const float* __restrict__ m,
                                               const float* __restrict__ Wf,
                                               const float* __restrict__ bf_,
                                               float* __restrict__ outp) {
  int id = blockIdx.x * 256 + threadIdx.x;     // B*768
  int b = id / EMB, n = id % EMB;
  const float* mr = m + (size_t)b * EMB;
  float s = bf_[n];
  for (int e = 0; e < EMB; e++) s += mr[e] * Wf[(size_t)e * EMB + n];
  outp[id] = s;
}

// ---------------- host driver ------------------------------------------------
extern "C" void kernel_launch(void* const* d_in, const int* in_sizes, int n_in,
                              void* d_out, int out_size, void* d_ws, size_t ws_size,
                              hipStream_t stream) {
  const float* x      = (const float*)d_in[0];
  const float* text   = (const float*)d_in[1];
  const float* Wpatch = (const float*)d_in[2];
  const float* bpatch = (const float*)d_in[3];
  const float* cls    = (const float*)d_in[4];
  const float* g1     = (const float*)d_in[5];
  const float* be1    = (const float*)d_in[6];
  const float* Wq     = (const float*)d_in[7];
  const float* bq     = (const float*)d_in[8];
  const float* Wk     = (const float*)d_in[9];
  const float* bk     = (const float*)d_in[10];
  const float* Wv     = (const float*)d_in[11];
  const float* bv     = (const float*)d_in[12];
  const float* Wp     = (const float*)d_in[13];
  const float* bp     = (const float*)d_in[14];
  const float* g2     = (const float*)d_in[15];
  const float* be2    = (const float*)d_in[16];
  const float* g3     = (const float*)d_in[17];
  const float* be3    = (const float*)d_in[18];
  const float* Wf     = (const float*)d_in[19];
  const float* bf_    = (const float*)d_in[20];
  float* out = (float*)d_out;

  char* ws = (char*)d_ws;
  const size_t S_T = (size_t)BATCH * NP * EMB * 4;   // 66,060,288 B
  const size_t S_H = (size_t)BATCH * NP * EMB * 2;   // 33,030,144 B
  const size_t WEL = (size_t)EMB * EMB;              // f16 elements per weight

  float*    t        = (float*)(ws);
  _Float16* h16      = (_Float16*)(ws + S_T);
  _Float16* w16      = (_Float16*)(ws + S_T + S_H);
  _Float16* wt_patch = w16 + 0 * WEL;
  _Float16* wt_q     = w16 + 1 * WEL;
  _Float16* wt_k     = w16 + 2 * WEL;
  _Float16* wt_v     = w16 + 3 * WEL;
  _Float16* wt_p     = w16 + 4 * WEL;
  char* big = ws + S_T + S_H + 5 * WEL * 2;
  _Float16* p16  = (_Float16*)big;                 // dead after patch GEMM
  _Float16* q16  = (_Float16*)big;                 // aliases p16 (safe)
  _Float16* k16  = (_Float16*)(big + S_H);
  _Float16* v16  = (_Float16*)(big + 2 * S_H);
  _Float16* ao16 = (_Float16*)(big + 3 * S_H);
  float*    o    = (float*)big;                    // aliases dead q16+k16
  float*    pooled = (float*)(big + 4 * S_H);
  float*    poolln = pooled + BATCH * EMB;

  // weights -> transposed f16
  k_w_t16<<<2304, 256, 0, stream>>>(Wpatch, wt_patch);
  k_w_t16<<<2304, 256, 0, stream>>>(Wq, wt_q);
  k_w_t16<<<2304, 256, 0, stream>>>(Wk, wt_k);
  k_w_t16<<<2304, 256, 0, stream>>>(Wv, wt_v);
  k_w_t16<<<2304, 256, 0, stream>>>(Wp, wt_p);

  k_im2col<<<(BATCH * NPATCH * EMB) / 256, 256, 0, stream>>>(x, p16);
  k_init_t<<<BATCH * NP, 256, 0, stream>>>(text, cls, t);

  // patch embed: t rows 1..576 per batch
  k_gemm<MODE_PATCH><<<dim3(144, 6), 256, 0, stream>>>(p16, wt_patch, bpatch, t,
                                                       BATCH * NPATCH);
  // LN1 -> h16 (f16, pad rows zero)
  k_ln1<<<BATCH * NP, 256, 0, stream>>>(t, g1, be1, h16);
  // QKV projections
  k_gemm<MODE_QKV><<<dim3(168, 6), 256, 0, stream>>>(h16, wt_q, bq, q16, BATCH * NP);
  k_gemm<MODE_QKV><<<dim3(168, 6), 256, 0, stream>>>(h16, wt_k, bk, k16, BATCH * NP);
  k_gemm<MODE_VT ><<<dim3(168, 6), 256, 0, stream>>>(h16, wt_v, bv, v16, BATCH * NP);
  // attention
  k_attn<<<dim3(6, BATCH * HEADS), 256, 0, stream>>>(q16, k16, v16, ao16);
  // output projection
  k_gemm<MODE_PLAIN><<<dim3(168, 6), 256, 0, stream>>>(ao16, wt_p, bp, o, BATCH * NP);
  // LN2 + residual into t
  k_ln2res<<<BATCH * NP, 256, 0, stream>>>(o, g2, be2, t);
  // mean pool -> LN3 -> final linear
  k_pool<<<(BATCH * EMB) / 256, 256, 0, stream>>>(t, pooled);
  k_lnrow<<<BATCH, 256, 0, stream>>>(pooled, g3, be3, poolln);
  k_final<<<(BATCH * EMB) / 256, 256, 0, stream>>>(poolln, Wf, bf_, out);
  (void)in_sizes; (void)n_in; (void)out_size; (void)ws_size;
}
